// CausalWindowAttention_33243046871750
// MI455X (gfx1250) — matrix operand order repaired and checked by reading, not compiled, not run on hardware
//
#include <hip/hip_runtime.h>

// ---------------------------------------------------------------------------
// Causal sliding-window attention (W=128) for MI455X (gfx1250, wave32, WMMA).
// D = A*B+C via v_wmma_f32_16x16x32_bf16. All GEMM-ish math in bf16 with f32
// accumulation.  Layout conventions (CDNA5 ISA 7.12.2, wave32):
//   A (16x32, bf16): lane holds row m = lane%16; element e -> k =
//                    (e<8 ? e : e+8) + 8*(lane/16)           (two b128 chunks)
//   B (32x16, bf16): lane holds col n = lane%16; element e -> k =
//                    (e<8 ? e : e+8) + 8*(lane/16), as A; V^T@P^T uses e + 16*(lane/16) for A and B
//   C/D (16x16,f32): lane holds col n = lane%16; vgpr v -> row m = v+8*(lane/16)
// ---------------------------------------------------------------------------

#ifndef __has_builtin
#define __has_builtin(x) 0
#endif

#if __has_builtin(__builtin_amdgcn_global_load_async_to_lds_b128)
#define HAVE_ASYNC_LDS 1
#else
#define HAVE_ASYNC_LDS 0
#endif

typedef __attribute__((ext_vector_type(16))) __bf16        v16bf;
typedef __attribute__((ext_vector_type(2)))  __bf16        v2bf;
typedef __attribute__((ext_vector_type(8)))  float         v8f;
typedef __attribute__((ext_vector_type(4)))  unsigned int  u32x4;
typedef __attribute__((ext_vector_type(4)))  int           v4i;

// Address-space-qualified v4i pointers for the async global->LDS builtin
// (param types per hipcc diagnostic: v4i in AS1 / AS3).
typedef __attribute__((address_space(1))) v4i* gas_v4i_ptr;
typedef __attribute__((address_space(3))) v4i* las_v4i_ptr;

union FragBF {
  v16bf v;
  u32x4 q[2];
  unsigned int u[8];
};

// Native bf16 converts: lowers to v_cvt_pk_bf16_f32 (RNE) on gfx1250.
static __device__ __forceinline__ unsigned short f2bf(float f) {
  __bf16 h = (__bf16)f;
  return __builtin_bit_cast(unsigned short, h);
}
static __device__ __forceinline__ unsigned int pack2bf(float a, float b) {
  v2bf t;
  t.x = (__bf16)a;
  t.y = (__bf16)b;
  return __builtin_bit_cast(unsigned int, t);
}

static __device__ __forceinline__ v8f wmma_bf16(const v16bf& a, const v16bf& b,
                                                const v8f& c) {
  return __builtin_amdgcn_wmma_f32_16x16x32_bf16(false, a, false, b, (short)0,
                                                 c, false, false);
}

// ---------------------------------------------------------------------------
// GEMM: C[16384 x 1024] = A[16384 x 1024] @ W[1024 x 1024] + bias
//   mode 0 (QKV): blockIdx.z selects (W,bias,out); output bf16 in [B,H,T,hd]
//   mode 1 (out-proj): uses slot-0 pointers; output f32 row-major [m,n]
// Block: 256 threads (8 waves).  Tile: 128(M) x 64(N), BK = 32.
// ---------------------------------------------------------------------------
__global__ __launch_bounds__(256) void gemm_bf16_wmma(
    const float* __restrict__ A,
    const float* __restrict__ W0, const float* __restrict__ W1,
    const float* __restrict__ W2,
    const float* __restrict__ b0, const float* __restrict__ b1,
    const float* __restrict__ b2,
    unsigned short* __restrict__ o0, unsigned short* __restrict__ o1,
    unsigned short* __restrict__ o2,
    float* __restrict__ of32, int mode) {
  __shared__ unsigned short As[128 * 32];  // [m][k] bf16, 8 KB
  __shared__ unsigned short Wt[64 * 32];   // [n][k] bf16 (transposed), 4 KB

  const int tid  = threadIdx.x;
  const int wave = tid >> 5;
  const int lane = tid & 31;
  const int ln   = lane & 15;
  const int hf   = lane >> 4;
  const int m0   = blockIdx.x * 128;
  const int n0   = blockIdx.y * 64;

  const float* Wm;
  const float* bias;
  unsigned short* obf = nullptr;
  if (mode == 0) {
    const int z = blockIdx.z;
    Wm   = (z == 0) ? W0 : (z == 1) ? W1 : W2;
    bias = (z == 0) ? b0 : (z == 1) ? b1 : b2;
    obf  = (z == 0) ? o0 : (z == 1) ? o1 : o2;
  } else {
    Wm   = W0;
    bias = b0;
  }

  v8f acc[4] = {};

  const int smm   = tid >> 1;         // A staging: row 0..127
  const int skoff = (tid & 1) * 16;   //            16 k-values each
  const int skk   = tid >> 3;         // W staging: k row 0..31
  const int sns   = (tid & 7) * 8;    //            8 contiguous n each

  for (int k0 = 0; k0 < 1024; k0 += 32) {
    {  // stage A tile: f32 -> bf16 pairs (v_cvt_pk_bf16_f32), 32-bit stores
      const float* ap = A + (m0 + smm) * 1024 + k0 + skoff;
#pragma unroll
      for (int j = 0; j < 8; ++j)
        *(unsigned int*)&As[smm * 32 + skoff + 2 * j] =
            pack2bf(ap[2 * j], ap[2 * j + 1]);
      if (k0 + 32 < 1024) __builtin_prefetch(ap + 32, 0, 1);  // next k-tile
    }
    {  // stage W tile transposed (coalesced read, transposed LDS write)
      const float* wp = Wm + (k0 + skk) * 1024 + n0 + sns;
#pragma unroll
      for (int j = 0; j < 8; ++j) Wt[(sns + j) * 32 + skk] = f2bf(wp[j]);
    }
    __syncthreads();

    FragBF a;
    const int ab = (wave * 16 + ln) * 32 + 8 * hf;
    a.q[0] = *(const u32x4*)&As[ab];
    a.q[1] = *(const u32x4*)&As[ab + 16];

#pragma unroll
    for (int nb = 0; nb < 4; ++nb) {
      FragBF bf;
      const int bb = (nb * 16 + ln) * 32 + 8 * hf;
      bf.q[0] = *(const u32x4*)&Wt[bb];
      bf.q[1] = *(const u32x4*)&Wt[bb + 16];
      acc[nb] = wmma_bf16(a.v, bf.v, acc[nb]);
    }
    __syncthreads();
  }

#pragma unroll
  for (int nb = 0; nb < 4; ++nb) {
    const int n  = n0 + nb * 16 + ln;
    const float bv = bias[n];
#pragma unroll
    for (int v = 0; v < 8; ++v) {
      const int m    = m0 + wave * 16 + v + 8 * hf;
      const float val = acc[nb][v] + bv;
      if (mode == 0) {
        const int bi = m >> 12, t = m & 4095;      // token -> (batch, time)
        const int hh = n >> 6, hd = n & 63;        // col   -> (head, dim)
        obf[(((bi * 16 + hh) * 4096) + t) * 64 + hd] = f2bf(val);
      } else {
        of32[m * 1024 + n] = val;
      }
    }
  }
}

// ---------------------------------------------------------------------------
// Banded attention, one workgroup per (query-block n, head, batch).
// Computes transposed:  S^T = K_win @ Q^T,  O^T = V^T @ P^T.
// Block: 256 threads (8 waves); wave w owns 16 queries, all 256 key slots.
// LDS: K[slot][hd] 32KB + V^T[hd][slot] 32KB = 64 KB.
// K staging uses gfx1250 async global->LDS copies when available.
// ---------------------------------------------------------------------------
__global__ __launch_bounds__(256) void attn_wmma(
    const unsigned short* __restrict__ Qb, const unsigned short* __restrict__ Kb,
    const unsigned short* __restrict__ Vb, float* __restrict__ ctx) {
  __shared__ unsigned short Kbuf[256 * 64];  // [slot][hd]
  __shared__ unsigned short Vt[64 * 256];    // [hd][slot]

  const int tid  = threadIdx.x;
  const int wave = tid >> 5;
  const int lane = tid & 31;
  const int ln   = lane & 15;
  const int hf   = lane >> 4;
  const int nblk = blockIdx.x;  // 0..31 query blocks of 128
  const int hh   = blockIdx.y;  // head
  const int b    = blockIdx.z;  // batch
  const long headbase = ((long)(b * 16 + hh)) * 4096 * 64;

  {  // stage: thread tid handles key slot = tid (tok = block start - W + slot)
    const int slot = tid;
    const int tok  = nblk * 128 - 128 + slot;
    if (tok >= 0) {
      const unsigned short* kp = Kb + headbase + (long)tok * 64;
#if HAVE_ASYNC_LDS
#pragma unroll
      for (int j = 0; j < 8; ++j)
        __builtin_amdgcn_global_load_async_to_lds_b128(
            (gas_v4i_ptr)(kp + j * 8),
            (las_v4i_ptr)&Kbuf[slot * 64 + j * 8],
            0, 0);
#else
#pragma unroll
      for (int j = 0; j < 8; ++j)
        *(u32x4*)&Kbuf[slot * 64 + j * 8] = *(const u32x4*)&kp[j * 8];
#endif
      const unsigned short* vp = Vb + headbase + (long)tok * 64;
#pragma unroll
      for (int j = 0; j < 64; ++j) Vt[j * 256 + slot] = vp[j];  // transpose
    } else {
      const u32x4 z = {0u, 0u, 0u, 0u};
#pragma unroll
      for (int j = 0; j < 8; ++j) *(u32x4*)&Kbuf[slot * 64 + j * 8] = z;
#pragma unroll
      for (int j = 0; j < 64; ++j) Vt[j * 256 + slot] = 0;
    }
  }
#if HAVE_ASYNC_LDS
#if __has_builtin(__builtin_amdgcn_s_wait_asynccnt)
  __builtin_amdgcn_s_wait_asynccnt(0);
#else
  asm volatile("s_wait_asynccnt 0x0" ::: "memory");
#endif
#endif
  __syncthreads();

  // ---- S^T = K @ Q^T : 16 slot-tiles x 2 k-steps over hd=64 ----
  v8f S[16] = {};
  const unsigned short* qrow =
      Qb + headbase + (long)(nblk * 128 + wave * 16 + ln) * 64;
#pragma unroll
  for (int kt = 0; kt < 2; ++kt) {
    FragBF qf;  // B operand: two 16-byte chunks straight from global
    qf.q[0] = *(const u32x4*)&qrow[kt * 32 + 8 * hf];
    qf.q[1] = *(const u32x4*)&qrow[kt * 32 + 8 * hf + 16];
#pragma unroll
    for (int st = 0; st < 16; ++st) {
      FragBF a;
      const int ab = (st * 16 + ln) * 64 + kt * 32 + 8 * hf;
      a.q[0] = *(const u32x4*)&Kbuf[ab];
      a.q[1] = *(const u32x4*)&Kbuf[ab + 16];
      S[st] = wmma_bf16(a.v, qf.v, S[st]);
    }
  }

  // ---- band mask + softmax over slots (pure registers + one half-shuffle) --
  const int r = wave * 16 + ln;  // query row within block
  float mx = -3.0e38f;
#pragma unroll
  for (int tt = 0; tt < 16; ++tt)
#pragma unroll
    for (int v = 0; v < 8; ++v) {
      const int slot = tt * 16 + v + 8 * hf;
      const bool valid =
          (slot >= r + 1) && (slot <= r + 128) && (nblk > 0 || slot >= 128);
      const float sc = valid ? S[tt][v] * 0.125f : -3.0e38f;  // /sqrt(64)
      S[tt][v] = sc;
      mx = fmaxf(mx, sc);
    }
  mx = fmaxf(mx, __shfl_xor(mx, 16, 32));
  float sum = 0.f;
#pragma unroll
  for (int tt = 0; tt < 16; ++tt)
#pragma unroll
    for (int v = 0; v < 8; ++v) {
      const float p = __expf(S[tt][v] - mx);
      S[tt][v] = p;
      sum += p;
    }
  sum += __shfl_xor(sum, 16, 32);
  const float inv = 1.0f / sum;

  // pack P^T rows pairwise into bf16x2 (per tile: 4 dwords = rows v, v+1)
  unsigned int pp[16][4];
#pragma unroll
  for (int tt = 0; tt < 16; ++tt)
#pragma unroll
    for (int j = 0; j < 4; ++j)
      pp[tt][j] = pack2bf(S[tt][2 * j] * inv, S[tt][2 * j + 1] * inv);

  // ---- O^T = V^T @ P^T : 4 hd-tiles x 8 k-steps over 256 slots ----
  v8f O[4] = {};
#pragma unroll
  for (int kt = 0; kt < 8; ++kt) {
    // B operand from accumulators: lane-half hf needs tile 2*kt+hf; rows 0..7
    // live in half-0 lanes, rows 8..15 in half-1 lanes -> shfl_xor(16) swap.
    FragBF pf;
#pragma unroll
    for (int j = 0; j < 4; ++j) {
      const unsigned int own0 = pp[2 * kt][j];
      const unsigned int own1 = pp[2 * kt + 1][j];
      const unsigned int oth0 = (unsigned int)__shfl_xor((int)own0, 16, 32);
      const unsigned int oth1 = (unsigned int)__shfl_xor((int)own1, 16, 32);
      pf.u[j]     = hf ? oth1 : own0;
      pf.u[j + 4] = hf ? own1 : oth0;
    }
#pragma unroll
    for (int mt = 0; mt < 4; ++mt) {
      FragBF a;
      const int ab = (mt * 16 + ln) * 256 + kt * 32 + 16 * hf;
      a.q[0] = *(const u32x4*)&Vt[ab];
      a.q[1] = *(const u32x4*)&Vt[ab + 8];
      O[mt] = wmma_bf16(a.v, pf.v, O[mt]);
    }
  }

  // ---- write context f32 [B,T,D] ----
  const int t = nblk * 128 + wave * 16 + ln;
  float* crow = ctx + ((long)(b * 4096 + t)) * 1024 + hh * 64;
#pragma unroll
  for (int mt = 0; mt < 4; ++mt)
#pragma unroll
    for (int v = 0; v < 8; ++v) crow[mt * 16 + v + 8 * hf] = O[mt][v];
}

// ---------------------------------------------------------------------------
extern "C" void kernel_launch(void* const* d_in, const int* in_sizes, int n_in,
                              void* d_out, int out_size, void* d_ws,
                              size_t ws_size, hipStream_t stream) {
  (void)in_sizes; (void)n_in; (void)out_size; (void)ws_size;
  const float* x  = (const float*)d_in[0];
  const float* Wq = (const float*)d_in[1];
  const float* bq = (const float*)d_in[2];
  const float* Wk = (const float*)d_in[3];
  const float* bk = (const float*)d_in[4];
  const float* Wv = (const float*)d_in[5];
  const float* bv = (const float*)d_in[6];
  const float* Wo = (const float*)d_in[7];
  const float* bo = (const float*)d_in[8];

  const long BTD = 4L * 4096 * 1024;  // 16,777,216
  unsigned short* qb = (unsigned short*)d_ws;  // bf16 [B,H,T,hd]
  unsigned short* kb = qb + BTD;
  unsigned short* vb = kb + BTD;               // needs 96 MB workspace

  float* out = (float*)d_out;        // output  [B,T,D]
  float* ctx = out + BTD;            // context [B,T,D]

  const dim3 blk(256);
  // 1) Q/K/V projections (+bias) -> bf16 head-major workspace
  gemm_bf16_wmma<<<dim3(128, 16, 3), blk, 0, stream>>>(
      x, Wq, Wk, Wv, bq, bk, bv, qb, kb, vb, nullptr, 0);
  // 2) banded attention -> f32 context (second output)
  attn_wmma<<<dim3(32, 16, 4), blk, 0, stream>>>(qb, kb, vb, ctx);
  // 3) out-projection (+bias) from f32 context -> f32 output (first output)
  gemm_bf16_wmma<<<dim3(128, 16, 1), blk, 0, stream>>>(
      ctx, Wo, Wo, Wo, bo, bo, bo, qb, qb, qb, out, 1);
}
